// Conv2d_15719580304127
// MI455X (gfx1250) — 8-bit matrix operand order repaired and checked by reading, compiled, not run on hardware
//
#include <hip/hip_runtime.h>
#include <hip/hip_bf16.h>

typedef __attribute__((ext_vector_type(8))) int   v8i;
typedef __attribute__((ext_vector_type(4))) int   v4ie;
typedef __attribute__((ext_vector_type(4))) float v4fe;
typedef int v4i_g __attribute__((vector_size(16)));

#define CB 32
#define CIN 64
#define COUT 256
#define HH 112
#define WW 112

#define WS_WQ   0
#define WS_DQ   147456
#define WS_BQ   148480
#define WS_X8   149504

#if __has_builtin(__builtin_amdgcn_global_load_async_to_lds_b128)
#define HAVE_ASYNC_LDS 1
#endif

__global__ __launch_bounds__(32) void k_quant_w(const float* __restrict__ w,
                                                const float* __restrict__ wscale,
                                                signed char* __restrict__ wqpack) {
    const int tap    = blockIdx.x >> 4;
    const int coTile = blockIdx.x & 15;
    const int lane   = threadIdx.x;
    const int kh = tap / 3, kw = tap % 3;
    const int co = coTile * 16 + (lane & 15);
    const float s = wscale[co];
    int pk[8];
#pragma unroll
    for (int v = 0; v < 8; ++v) {
        int word = 0;
#pragma unroll
        for (int j = 0; j < 4; ++j) {
            const int cin = ((v & 3) << 2) + j + ((lane >> 4) << 4) + ((v >> 2) << 5);
            const float wv = w[((co * CIN + cin) * 3 + kh) * 3 + kw];
            float q = rintf(wv / s);
            q = fminf(fmaxf(q, -128.0f), 127.0f);
            word |= (((int)q) & 255) << (8 * j);
        }
        pk[v] = word;
    }
    int4* dst = (int4*)(wqpack + ((tap * 16 + coTile) * 32 + lane) * 32);
    dst[0] = make_int4(pk[0], pk[1], pk[2], pk[3]);
    dst[1] = make_int4(pk[4], pk[5], pk[6], pk[7]);
}

__global__ __launch_bounds__(256) void k_quant_bias(const float* __restrict__ bias,
                                                    const float* __restrict__ wscale,
                                                    const float* __restrict__ in_scale,
                                                    float* __restrict__ dq,
                                                    float* __restrict__ bq) {
    const int c = threadIdx.x;
    const float d = in_scale[0] * wscale[c];
    dq[c] = d;
    bq[c] = rintf(bias[c] / d);
}

__global__ __launch_bounds__(256) void k_pack_x(const int* __restrict__ x,
                                                signed char* __restrict__ x8) {
    __shared__ __align__(16) signed char tile[64 * 64];
    const int t  = threadIdx.x;
    const int wt = blockIdx.x & 1;
    const int h  = (blockIdx.x >> 1) % HH;
    const int nb = blockIdx.x / (2 * HH);
    const int w0 = wt * 64;
    const int wL = t & 63;
    const int crow = t >> 6;
    const int w = w0 + wL;
    if (w < WW) {
#pragma unroll
        for (int i = 0; i < 16; ++i) {
            const int c = crow * 16 + i;
            const int v = __builtin_nontemporal_load(x + ((nb * CIN + c) * HH + h) * WW + w);
            tile[wL * 64 + c] = (signed char)v;
        }
    }
    __syncthreads();
#pragma unroll
    for (int i = 0; i < 4; ++i) {
        const int idx = i * 256 + t;
        const int pix = idx >> 4;
        const int dw  = idx & 15;
        const int d2  = dw & 7;
        const int cbase = ((dw >> 3) << 4) + ((d2 & 3) << 2) + ((d2 >> 2) << 5);
        if (w0 + pix < WW) {
            const int val = *(const int*)(tile + pix * 64 + cbase);
            *(int*)(x8 + ((nb * HH + h) * WW + w0 + pix) * 64 + dw * 4) = val;
        }
    }
}

__device__ __forceinline__ void stage16k(const signed char* gsrc, signed char* ldst,
                                         int t) {
#ifdef HAVE_ASYNC_LDS
#pragma unroll
    for (int i = 0; i < 4; ++i) {
        const int off = (i * 256 + t) * 16;
        __builtin_amdgcn_global_load_async_to_lds_b128(
            (v4i_g*)(gsrc + off), (v4i_g*)(ldst + off), 0, 0);
    }
#else
#pragma unroll
    for (int i = 0; i < 4; ++i) {
        const int off = (i * 256 + t) * 16;
        *(int4*)(ldst + off) = *(const int4*)(gsrc + off);
    }
#endif
}

__device__ __forceinline__ void wait_async0() {
#if __has_builtin(__builtin_amdgcn_s_wait_asynccnt)
    __builtin_amdgcn_s_wait_asynccnt(0);
#else
    asm volatile("s_wait_asynccnt 0x0" ::: "memory");
#endif
}

__global__ __launch_bounds__(256) void k_conv(const signed char* __restrict__ x8,
                                              const signed char* __restrict__ wq,
                                              const float* __restrict__ dq,
                                              const float* __restrict__ bq,
                                              const int* __restrict__ sumin,
                                              const float* __restrict__ sumin_scale,
                                              const float* __restrict__ out_scale,
                                              float* __restrict__ out) {
    __shared__ __align__(16) signed char ldsW[2][16384];
    const int t    = threadIdx.x;
    const int lane = t & 31;
    const int wave = t >> 5;
    const int hi   = lane >> 4;
    const int ln   = lane & 15;
    const int coHalf = wave & 1;
    const int T = blockIdx.x * 4 + (wave >> 1);
    const int wt = T % 7;
    const int h  = (T / 7) % HH;
    const int nb = T / (7 * HH);
    const int w0 = wt * 16;

#pragma unroll
    for (int ct = 0; ct < 8; ++ct) {
        const int co = (coHalf * 8 + ct) * 16 + ln;
        __builtin_prefetch(sumin + ((nb * COUT + co) * HH + h) * WW + w0 + hi * 8, 0, 3);
    }

    v8i zero = {0, 0, 0, 0, 0, 0, 0, 0};
    v8i acc[8];
#pragma unroll
    for (int ct = 0; ct < 8; ++ct) acc[ct] = zero;

    stage16k(wq, ldsW[0], t);
    wait_async0();
    __syncthreads();

#pragma unroll
    for (int tap = 0; tap < 9; ++tap) {
        if (tap < 8) stage16k(wq + (tap + 1) * 16384, ldsW[(tap + 1) & 1], t);
        const signed char* ldsBuf = ldsW[tap & 1];

        const int kh = tap / 3 - 1;
        const int kw = tap % 3 - 1;
        const int hh = h + kh;
        const int ww = w0 + ln + kw;
        const bool valid = ((unsigned)hh < HH) && ((unsigned)ww < WW);
        int4 q0 = make_int4(0, 0, 0, 0), q1 = make_int4(0, 0, 0, 0);
        if (valid) {
            const signed char* ap = x8 + (((nb * HH + hh) * WW + ww) << 6) + (hi << 5);
            q0 = *(const int4*)(ap);
            q1 = *(const int4*)(ap + 16);
        }
        v8i a;
        a[0] = q0.x; a[1] = q0.y; a[2] = q0.z; a[3] = q0.w;
        a[4] = q1.x; a[5] = q1.y; a[6] = q1.z; a[7] = q1.w;

#pragma unroll
        for (int ct = 0; ct < 8; ++ct) {
            const int coTile = coHalf * 8 + ct;
            const int4* bsrc = (const int4*)(ldsBuf + (coTile * 32 + lane) * 32);
            const int4 b0 = bsrc[0];
            const int4 b1 = bsrc[1];
            v8i bf;
            bf[0] = b0.x; bf[1] = b0.y; bf[2] = b0.z; bf[3] = b0.w;
            bf[4] = b1.x; bf[5] = b1.y; bf[6] = b1.z; bf[7] = b1.w;
            acc[ct] = __builtin_amdgcn_wmma_i32_16x16x64_iu8(
                true, a, true, bf, acc[ct], false, false);
        }

        wait_async0();
        __syncthreads();
    }

    const float ss = sumin_scale[0];
    const float inv_os = 1.0f / out_scale[0];
#pragma unroll
    for (int ct = 0; ct < 8; ++ct) {
        const int coTile = coHalf * 8 + ct;
        const int co = coTile * 16 + ln;
        const float d  = dq[co];
        const float bb = bq[co];
        const int base = ((nb * COUT + co) * HH + h) * WW + w0 + hi * 8;
        const v4ie s0 = __builtin_nontemporal_load((const v4ie*)(sumin + base));
        const v4ie s1 = __builtin_nontemporal_load((const v4ie*)(sumin + base + 4));
        const int sv[8] = {s0.x, s0.y, s0.z, s0.w, s1.x, s1.y, s1.z, s1.w};
        float r[8];
#pragma unroll
        for (int i = 0; i < 8; ++i) {
            float af = ((float)acc[ct][i] + bb) * d + (float)sv[i] * ss;
            af = fmaxf(af, 0.0f);
            float q = rintf(af * inv_os);
            r[i] = fminf(fmaxf(q, -128.0f), 127.0f);
        }
        v4fe o0 = {r[0], r[1], r[2], r[3]};
        v4fe o1 = {r[4], r[5], r[6], r[7]};
        __builtin_nontemporal_store(o0, (v4fe*)(out + base));
        __builtin_nontemporal_store(o1, (v4fe*)(out + base + 4));
    }
}

extern "C" void kernel_launch(void* const* d_in, const int* in_sizes, int n_in,
                              void* d_out, int out_size, void* d_ws, size_t ws_size,
                              hipStream_t stream) {
    const int*   x        = (const int*)d_in[0];
    const int*   sumin    = (const int*)d_in[1];
    const float* weight   = (const float*)d_in[2];
    const float* bias     = (const float*)d_in[3];
    const float* wscale   = (const float*)d_in[4];
    const float* in_scale = (const float*)d_in[5];
    const float* ss       = (const float*)d_in[6];
    const float* os       = (const float*)d_in[7];
    float* out = (float*)d_out;

    char* ws = (char*)d_ws;
    signed char* wqpack = (signed char*)(ws + WS_WQ);
    float*       dq     = (float*)(ws + WS_DQ);
    float*       bq     = (float*)(ws + WS_BQ);
    signed char* x8     = (signed char*)(ws + WS_X8);

    k_quant_w<<<9 * 16, 32, 0, stream>>>(weight, wscale, wqpack);
    k_quant_bias<<<1, 256, 0, stream>>>(bias, wscale, in_scale, dq, bq);
    k_pack_x<<<CB * HH * 2, 256, 0, stream>>>(x, x8);
    k_conv<<<25088 / 4, 256, 0, stream>>>(x8, wqpack, dq, bq, sumin, ss, os, out);
}
